// SemanticMultiHeadAttention_86801289052597
// MI455X (gfx1250) — matrix operand order repaired and checked by reading, not compiled, not run on hardware
//
#include <hip/hip_runtime.h>
#include <hip/hip_bf16.h>

#define B_  8
#define S_  1024
#define D_  1024
#define NH_ 16
#define DH_ 64

typedef __attribute__((ext_vector_type(16))) __bf16 v16bf;
typedef __attribute__((ext_vector_type(8)))  float  v8f;
typedef __attribute__((ext_vector_type(4)))  float  v4f;
typedef __attribute__((ext_vector_type(4)))  unsigned int v4u;   // 16 bytes

union AFrag { v16bf v; __bf16 h[16]; unsigned short u[16]; v4u q[2]; };
union CFrag { v8f v; float f[8]; };

__device__ __forceinline__ unsigned short f2bfu(float f) {
    __bf16 h = (__bf16)f;                 // v_cvt_pk_bf16_f32 on gfx1250
    return __builtin_bit_cast(unsigned short, h);
}

// ---- WMMA bf16 fragment loaders (wave32 layouts per CDNA5 ISA 7.12.2) ----
// A (16x32, MxK): lane L<16 -> M=L, u[0..7]=K(khalf+0..7), u[8..15]=K(16+khalf..);
//                 lane L>=16 -> khalf = 8.  Contiguous along K -> b128 loads.
__device__ __forceinline__ v16bf load_a_bf16(const unsigned short* src, int ld) {
    int lane = threadIdx.x & 31;
    int m = lane & 15;
    int khalf = (lane >> 4) * 8;
    const unsigned short* p = src + (size_t)m * ld + khalf;
    AFrag r;
    r.q[0] = *(const v4u*)p;            // K = khalf .. khalf+7
    r.q[1] = *(const v4u*)(p + 16);     // K = 16+khalf .. 16+khalf+7
    return r.v;
}
__device__ __forceinline__ v16bf load_a_f32(const float* src, int ld) {
    int lane = threadIdx.x & 31;
    int m = lane & 15;
    int khalf = (lane >> 4) * 8;
    const float* p = src + (size_t)m * ld + khalf;
    v4f x0 = *(const v4f*)p;
    v4f x1 = *(const v4f*)(p + 4);
    v4f x2 = *(const v4f*)(p + 16);
    v4f x3 = *(const v4f*)(p + 20);
    AFrag r;
#pragma unroll
    for (int i = 0; i < 4; ++i) {
        r.h[i]      = (__bf16)x0[i];
        r.h[4 + i]  = (__bf16)x1[i];
        r.h[8 + i]  = (__bf16)x2[i];
        r.h[12 + i] = (__bf16)x3[i];
    }
    return r.v;
}
// B (32x16, KxN): lane L<16 -> N=L, u[0..7]=K(kb+0..7), u[8..15]=K(16+kb..); lane>=16 -> kb=8.
// n-major source: element (k,n) at src[n*ld + k]  -> contiguous along k.
__device__ __forceinline__ v16bf load_b_bf16_nmaj(const unsigned short* src, int ld) {
    int lane = threadIdx.x & 31;
    int n = lane & 15;
    int kb = (lane >> 4) * 8;
    const unsigned short* p = src + (size_t)n * ld + kb;
    AFrag r;
    r.q[0] = *(const v4u*)p;
    r.q[1] = *(const v4u*)(p + 16);
    return r.v;
}
// C/D (16x16 f32): lane L<16 -> N=L, rows 0..7; lane>=16 -> N=L-16, rows 8..15.
__device__ __forceinline__ void store_c_f32(float* dst, int ld, v8f c) {
    int lane = threadIdx.x & 31;
    int n = lane & 15;
    int rb = (lane >> 4) * 8;
    CFrag u; u.v = c;
#pragma unroll
    for (int i = 0; i < 8; ++i) dst[(size_t)(rb + i) * ld + n] = u.f[i];
}
__device__ __forceinline__ void store_c_bf16(unsigned short* dst, int ld, v8f c) {
    int lane = threadIdx.x & 31;
    int n = lane & 15;
    int rb = (lane >> 4) * 8;
    CFrag u; u.v = c;
#pragma unroll
    for (int i = 0; i < 8; ++i) dst[(size_t)(rb + i) * ld + n] = f2bfu(u.f[i]);
}
// Transposed bf16 store: element (row,col) -> dst[col*ld + row].
// Per lane the 8 rows are contiguous -> single b128 store.
__device__ __forceinline__ void store_c_bf16_T(unsigned short* dst, int ld, v8f c) {
    int lane = threadIdx.x & 31;
    int n = lane & 15;
    int rb = (lane >> 4) * 8;
    CFrag u; u.v = c;
    union { unsigned short s[8]; v4u q; } t;
#pragma unroll
    for (int i = 0; i < 8; ++i) t.s[i] = f2bfu(u.f[i]);
    *(v4u*)(dst + (size_t)n * ld + rb) = t.q;
}

__device__ __forceinline__ v8f wmma_bf16(v16bf a, v16bf b, v8f c) {
    return __builtin_amdgcn_wmma_f32_16x16x32_bf16(false, a, false, b, (short)0, c,
                                                   false, false);
}

// ---------------- K0: weights f32 -> bf16 (once) ----------------
__global__ void k_cvt_w(const float* __restrict__ Wo, const float* __restrict__ Wq,
                        const float* __restrict__ Wk, const float* __restrict__ Wv,
                        unsigned short* __restrict__ Wob, unsigned short* __restrict__ Wqb,
                        unsigned short* __restrict__ Wkb, unsigned short* __restrict__ Wvb) {
    size_t i = (size_t)blockIdx.x * 256 + threadIdx.x;
    const size_t NW = (size_t)D_ * D_;            // 1M
    const size_t NP = (size_t)NH_ * DH_ * DH_;    // 64K
    if (i < NW) Wob[i] = f2bfu(Wo[i]);
    if (i < NP) {
        Wqb[i] = f2bfu(Wq[i]);
        Wkb[i] = f2bfu(Wk[i]);
        Wvb[i] = f2bfu(Wv[i]);
    }
}

// ---------------- K1: per-head Q/K/V projections ----------------
// out[s][e] = sum_d H[s][d] * W[e][d] + bias[e]
__device__ __forceinline__ void proj_one(v16bf a0, v16bf a1,
                                         const unsigned short* Wn, const float* biasN,
                                         unsigned short* dst, bool transpose, int sT,
                                         int ldT) {
    int lane = threadIdx.x & 31;
    int ncol = lane & 15;
    float bias[4];
#pragma unroll
    for (int et = 0; et < 4; ++et) bias[et] = biasN[et * 16 + ncol];
#pragma unroll
    for (int et = 0; et < 4; ++et) {
        int e0 = et * 16;
        CFrag c;
#pragma unroll
        for (int i = 0; i < 8; ++i) c.f[i] = bias[et];
        // B[k=d][n=e] = Wn[e][d]  (n-major, ld = DH)
        v16bf b0 = load_b_bf16_nmaj(Wn + (size_t)e0 * DH_, DH_);
        v16bf b1 = load_b_bf16_nmaj(Wn + (size_t)e0 * DH_ + 32, DH_);
        c.v = wmma_bf16(a0, b0, c.v);
        c.v = wmma_bf16(a1, b1, c.v);
        if (!transpose)
            store_c_bf16(dst + e0, DH_, c.v);               // [s][e], rows at dst
        else
            store_c_bf16_T(dst + (size_t)e0 * ldT + sT, ldT, c.v);  // [e][s]
    }
}

__global__ void __launch_bounds__(32, 1)
k_qkv(const float* __restrict__ H,
      const unsigned short* __restrict__ Wqb, const float* __restrict__ bq,
      const unsigned short* __restrict__ Wkb, const float* __restrict__ bk,
      const unsigned short* __restrict__ Wvb, const float* __restrict__ bv,
      unsigned short* __restrict__ Qb, unsigned short* __restrict__ Kb,
      unsigned short* __restrict__ Vt) {
    int b = blockIdx.x / NH_, n = blockIdx.x % NH_;
    int sT = blockIdx.y * 16;
    const float* Asrc = H + ((size_t)(b * S_ + sT)) * D_ + n * DH_;
    v16bf a0 = load_a_f32(Asrc, D_);        // d = 0..31
    v16bf a1 = load_a_f32(Asrc + 32, D_);   // d = 32..63
    size_t obase = ((size_t)(b * NH_ + n) * S_ + sT) * DH_;
    size_t tbase = (size_t)(b * NH_ + n) * DH_ * S_;
    proj_one(a0, a1, Wqb + (size_t)n * DH_ * DH_, bq + n * DH_, Qb + obase, false, 0, 0);
    proj_one(a0, a1, Wkb + (size_t)n * DH_ * DH_, bk + n * DH_, Kb + obase, false, 0, 0);
    proj_one(a0, a1, Wvb + (size_t)n * DH_ * DH_, bv + n * DH_, Vt + tbase, true, sT, S_);
}

// ---------------- K2: scores = QK^T/sqrt(DH), masked -> Aout (pre-softmax) -------
__global__ void __launch_bounds__(32, 1)
k_scores(const unsigned short* __restrict__ Qb,
         const unsigned short* __restrict__ Kb,
         const int* __restrict__ mask, float* __restrict__ Aout) {
    int n = blockIdx.x / B_, b = blockIdx.x % B_;
    int sT = blockIdx.y * 16;
    int tB = blockIdx.z * 64;
    const unsigned short* Qp = Qb + ((size_t)(b * NH_ + n) * S_ + sT) * DH_;
    v16bf a0 = load_a_bf16(Qp, DH_);
    v16bf a1 = load_a_bf16(Qp + 32, DH_);
    const unsigned short* Kp = Kb + (size_t)(b * NH_ + n) * S_ * DH_;
    int lane = threadIdx.x & 31;
    int nn = lane & 15;
    int rb = (lane >> 4) * 8;
#pragma unroll
    for (int tt = 0; tt < 4; ++tt) {
        int t0 = tB + tt * 16;
        // B[k=e][n=t] = K[t][e]  (n-major over t, ld = DH)
        v16bf b0 = load_b_bf16_nmaj(Kp + (size_t)t0 * DH_, DH_);
        v16bf b1 = load_b_bf16_nmaj(Kp + (size_t)t0 * DH_ + 32, DH_);
        CFrag c;
#pragma unroll
        for (int i = 0; i < 8; ++i) c.f[i] = 0.0f;
        c.v = wmma_bf16(a0, b0, c.v);
        c.v = wmma_bf16(a1, b1, c.v);
        int tcol = t0 + nn;
#pragma unroll
        for (int i = 0; i < 8; ++i) {
            int row = sT + rb + i;
            int mv = mask[((size_t)b * S_ + row) * S_ + tcol];
            float v = (mv == 0) ? -1e9f : c.f[i] * 0.125f;  // 1/sqrt(64)
            Aout[(((size_t)n * B_ + b) * S_ + row) * S_ + tcol] = v;
        }
    }
}

// ---------------- K3: row softmax over length S ----------------
__global__ void k_softmax(float* __restrict__ Aout) {
    __shared__ float red[256];
    float* p = Aout + (size_t)blockIdx.x * S_;
    int tid = threadIdx.x;
    v4f v = ((const v4f*)p)[tid];
    float m = fmaxf(fmaxf(v[0], v[1]), fmaxf(v[2], v[3]));
    red[tid] = m;
    __syncthreads();
    for (int s = 128; s > 0; s >>= 1) {
        if (tid < s) red[tid] = fmaxf(red[tid], red[tid + s]);
        __syncthreads();
    }
    m = red[0];
    __syncthreads();
    float sum = 0.0f;
#pragma unroll
    for (int i = 0; i < 4; ++i) { v[i] = __expf(v[i] - m); sum += v[i]; }
    red[tid] = sum;
    __syncthreads();
    for (int s = 128; s > 0; s >>= 1) {
        if (tid < s) red[tid] += red[tid + s];
        __syncthreads();
    }
    float inv = 1.0f / red[0];
#pragma unroll
    for (int i = 0; i < 4; ++i) v[i] *= inv;
    ((v4f*)p)[tid] = v;
}

// ---------------- K4: heads = A @ V -> bf16 concat ----------------
__global__ void __launch_bounds__(32, 1)
k_av(const float* __restrict__ Aout, const unsigned short* __restrict__ Vt,
     unsigned short* __restrict__ Cc) {
    int b = blockIdx.x / NH_, n = blockIdx.x % NH_;
    int sT = blockIdx.y * 16;
    const float* Ap = Aout + (((size_t)n * B_ + b) * S_ + sT) * S_;
    const unsigned short* Vp = Vt + (size_t)(b * NH_ + n) * DH_ * S_;  // [dh][t]
    CFrag c[4];
#pragma unroll
    for (int dt = 0; dt < 4; ++dt)
#pragma unroll
        for (int i = 0; i < 8; ++i) c[dt].f[i] = 0.0f;
    for (int kt = 0; kt < S_ / 32; ++kt) {
        v16bf a = load_a_f32(Ap + kt * 32, S_);
#pragma unroll
        for (int dt = 0; dt < 4; ++dt) {
            // B[k=t][n=dh] = Vt[dh][t]  (n-major over dh, ld = S)
            v16bf bb = load_b_bf16_nmaj(Vp + (size_t)dt * 16 * S_ + kt * 32, S_);
            c[dt].v = wmma_bf16(a, bb, c[dt].v);
        }
    }
    unsigned short* dst = Cc + (size_t)(b * S_ + sT) * D_ + n * DH_;
#pragma unroll
    for (int dt = 0; dt < 4; ++dt) store_c_bf16(dst + dt * 16, D_, c[dt].v);
}

// ---------------- K5: output = concat @ Wo^T + bo ----------------
__global__ void __launch_bounds__(32, 1)
k_outproj(const unsigned short* __restrict__ Cc,
          const unsigned short* __restrict__ Wob,
          const float* __restrict__ bo, float* __restrict__ Out) {
    int rT = blockIdx.x * 16;   // rows over B*S
    int jB = blockIdx.y * 64;
    const unsigned short* Ap = Cc + (size_t)rT * D_;
    int lane = threadIdx.x & 31;
    int nn = lane & 15;
    CFrag c[4];
#pragma unroll
    for (int jt = 0; jt < 4; ++jt) {
        float bias = bo[jB + jt * 16 + nn];
#pragma unroll
        for (int i = 0; i < 8; ++i) c[jt].f[i] = bias;
    }
    for (int kt = 0; kt < D_ / 32; ++kt) {
        v16bf a = load_a_bf16(Ap + kt * 32, D_);
        if (kt + 1 < D_ / 32)
            __builtin_prefetch(Ap + (kt + 1) * 32, 0, 1);
#pragma unroll
        for (int jt = 0; jt < 4; ++jt) {
            // B[k=i][n=j] = Wo[j][i]  (n-major over j, ld = D)
            v16bf bb = load_b_bf16_nmaj(Wob + (size_t)(jB + jt * 16) * D_ + kt * 32, D_);
            c[jt].v = wmma_bf16(a, bb, c[jt].v);
        }
    }
    float* dst = Out + (size_t)rT * D_ + jB;
#pragma unroll
    for (int jt = 0; jt < 4; ++jt) store_c_f32(dst + jt * 16, D_, c[jt].v);
}

extern "C" void kernel_launch(void* const* d_in, const int* in_sizes, int n_in,
                              void* d_out, int out_size, void* d_ws, size_t ws_size,
                              hipStream_t stream) {
    const float* H    = (const float*)d_in[0];
    const int*   mask = (const int*)d_in[1];
    const float* Wq   = (const float*)d_in[2];
    const float* bq   = (const float*)d_in[3];
    const float* Wk   = (const float*)d_in[4];
    const float* bk   = (const float*)d_in[5];
    const float* Wv   = (const float*)d_in[6];
    const float* bv   = (const float*)d_in[7];
    const float* Wo   = (const float*)d_in[8];
    const float* bo   = (const float*)d_in[9];

    float* Out  = (float*)d_out;                       // (B,S,D)
    float* Aout = Out + (size_t)B_ * S_ * D_;          // (NH,B,S,S)

    const size_t QKV = (size_t)B_ * NH_ * S_ * DH_;    // 8.4M bf16 elems each
    unsigned short* Qb  = (unsigned short*)d_ws;
    unsigned short* Kb  = Qb + QKV;
    unsigned short* Vt  = Kb + QKV;                    // V transposed: [b][n][dh][t]
    unsigned short* Cc  = Vt + QKV;                    // (B,S,D) bf16
    unsigned short* Wob = Cc + (size_t)B_ * S_ * D_;   // (D,D) bf16
    unsigned short* Wqb = Wob + (size_t)D_ * D_;       // (NH,DH,DH) bf16
    unsigned short* Wkb = Wqb + (size_t)NH_ * DH_ * DH_;
    unsigned short* Wvb = Wkb + (size_t)NH_ * DH_ * DH_;

    k_cvt_w<<<(D_ * D_ + 255) / 256, 256, 0, stream>>>(Wo, Wq, Wk, Wv,
                                                       Wob, Wqb, Wkb, Wvb);
    k_qkv<<<dim3(B_ * NH_, S_ / 16), 32, 0, stream>>>(H, Wqb, bq, Wkb, bk, Wvb, bv,
                                                      Qb, Kb, Vt);
    k_scores<<<dim3(NH_ * B_, S_ / 16, S_ / 64), 32, 0, stream>>>(Qb, Kb, mask, Aout);
    k_softmax<<<NH_ * B_ * S_, 256, 0, stream>>>(Aout);
    k_av<<<dim3(B_ * NH_, S_ / 16), 32, 0, stream>>>(Aout, Vt, Cc);
    k_outproj<<<dim3((B_ * S_) / 16, D_ / 64), 32, 0, stream>>>(Cc, Wob, bo, Out);
}
